// CrossModalAttention_26946624815689
// MI455X (gfx1250) — compile-verified, matrix operand order checked by reading, not run on hardware
//
#include <hip/hip_runtime.h>
#include <hip/hip_bf16.h>
#include <stdint.h>

#define B_ROWS 131072
#define E 256
#define M_TILE 64
#define XS_STRIDE (E + 8)   // pad 8 bf16 (16B) -> conflict-free b128 LDS reads

typedef __attribute__((ext_vector_type(16))) __bf16 v16bf;
typedef __attribute__((ext_vector_type(8)))  float  v8f;
typedef unsigned int u32x4 __attribute__((ext_vector_type(4)));

union FragU {
    u32x4 q[2];   // two 16B chunks = 16 bf16
    v16bf v;
};

__device__ __forceinline__ unsigned short f32_to_bf16_rne(float f) {
    union { float f; uint32_t u; } c; c.f = f;
    uint32_t u = c.u;
    uint32_t r = u + 0x7FFFu + ((u >> 16) & 1u);
    return (unsigned short)(r >> 16);
}

// Map (lane, j) of a WMMA bf16 operand fragment to k-in-tile.
// 16-bit A-matrix 16x32 layout (ISA 7.12.2): lanes 0-15 K=0..7,16..23;
// lanes 16-31 K=8..15,24..31; element j -> VGPR j/2 half j%2.
__device__ __forceinline__ int frag_k_off(int lane, int j) {
    return (j & 7) + ((j >= 8) ? 16 : 0) + ((lane >> 4) << 3);
}

// ---------------------------------------------------------------------------
// Precompute fused weight W = out_w @ Wv  (att = x @ Wv.T @ out_w.T = x @ W.T)
// packed directly in WMMA B-fragment order: for fragment f = n_tile*8+k_tile,
// packedW[((p*128+f)*32 + lane)*16 + j] = bf16( W[n][k] ) with the lane/j map.
// ---------------------------------------------------------------------------
__global__ void pack_fused_w(const float* __restrict__ a2t_in_w,
                             const float* __restrict__ a2t_out_w,
                             const float* __restrict__ t2a_in_w,
                             const float* __restrict__ t2a_out_w,
                             unsigned short* __restrict__ packedW) {
    int t = blockIdx.x * blockDim.x + threadIdx.x;   // 0..131071
    int p      = t >> 16;
    int rem    = t & 65535;
    int frag   = rem >> 9;        // 0..127
    int within = rem & 511;
    int lane   = within >> 4;
    int j      = within & 15;
    int n_tile = frag >> 3;
    int k_tile = frag & 7;

    int n = n_tile * 16 + (lane & 15);
    int k = k_tile * 32 + frag_k_off(lane, j);

    const float* in_w  = p ? t2a_in_w  : a2t_in_w;
    const float* out_w = p ? t2a_out_w : a2t_out_w;

    float acc = 0.f;
    for (int tt = 0; tt < E; ++tt)
        acc += out_w[n * E + tt] * in_w[(2 * E + tt) * E + k];

    packedW[((size_t)(p * 128 + frag) * 32 + lane) * 16 + j] = f32_to_bf16_rne(acc);
}

// fused bias: fb[n] = out_b[n] + sum_t out_w[n][t] * bv[t]
__global__ void fused_bias(const float* __restrict__ a2t_in_b,
                           const float* __restrict__ a2t_out_w,
                           const float* __restrict__ a2t_out_b,
                           const float* __restrict__ t2a_in_b,
                           const float* __restrict__ t2a_out_w,
                           const float* __restrict__ t2a_out_b,
                           float* __restrict__ fbias) {
    int t = blockIdx.x * blockDim.x + threadIdx.x;
    if (t >= 2 * E) return;
    int p = t >> 8, n = t & 255;
    const float* in_b  = p ? t2a_in_b  : a2t_in_b;
    const float* out_w = p ? t2a_out_w : a2t_out_w;
    const float* out_b = p ? t2a_out_b : a2t_out_b;
    float acc = out_b[n];
    for (int tt = 0; tt < E; ++tt)
        acc += out_w[n * E + tt] * in_b[2 * E + tt];
    fbias[t] = acc;
}

// ---------------------------------------------------------------------------
// Main fused kernel: BOTH paths per block over the same 64-row slab, so each
// embed tile is pulled from HBM once and all re-reads (residual / re-stage)
// hit L2. Per path: bf16 WMMA att-GEMM + residual + LayerNorm.
// grid = (B/64), block = 256 (8 waves). Wave w owns N-tiles {2w, 2w+1}.
// ---------------------------------------------------------------------------
__global__ __launch_bounds__(256) void cross_attn_main(
    const float* __restrict__ audio, const float* __restrict__ text,
    const unsigned short* __restrict__ packedW, const float* __restrict__ fbias,
    const float* __restrict__ ln1_g, const float* __restrict__ ln1_b,
    const float* __restrict__ ln2_g, const float* __restrict__ ln2_b,
    float* __restrict__ out)
{
    __shared__ __align__(16) unsigned short xs[M_TILE * XS_STRIDE]; // bf16 kv tile
    __shared__ float s_sum[M_TILE];
    __shared__ float s_ssq[M_TILE];

    const int m0  = blockIdx.x * M_TILE;
    const int tid = threadIdx.x;

    const int wave   = tid >> 5;
    const int lane   = tid & 31;
    const int lane16 = lane & 15;
    const int grp    = lane >> 4;
    const int nt0    = wave * 2;

    for (int p = 0; p < 2; ++p) {
        const float* src = p ? audio : text;   // kv input (produces att)
        const float* res = p ? text  : audio;  // residual
        const float* gam = p ? ln2_g : ln1_g;
        const float* bet = p ? ln2_b : ln1_b;

        if (tid < M_TILE)          s_sum[tid] = 0.f;
        else if (tid < 2 * M_TILE) s_ssq[tid - M_TILE] = 0.f;

        // stage kv tile as bf16 into LDS (coalesced f32 reads; path 1 is L2-hot)
        #pragma unroll
        for (int i = 0; i < (M_TILE * E) / 256; ++i) {
            int flat = i * 256 + tid;
            int row = flat >> 8, col = flat & 255;
            xs[row * XS_STRIDE + col] =
                f32_to_bf16_rne(src[(size_t)(m0 + row) * E + col]);
        }
        __syncthreads();

        // B fragments: 2 n_tiles x 8 k_tiles (L2-resident packed weights)
        FragU bfr[2][8];
        #pragma unroll
        for (int ni = 0; ni < 2; ++ni)
            #pragma unroll
            for (int k = 0; k < 8; ++k) {
                int frag = (nt0 + ni) * 8 + k;
                const u32x4* s4 = (const u32x4*)(packedW +
                    ((size_t)(p * 128 + frag) * 32 + lane) * 16);
                bfr[ni][k].q[0] = s4[0];
                bfr[ni][k].q[1] = s4[1];
            }

        float fb[2], gg[2], bb[2];
        #pragma unroll
        for (int ni = 0; ni < 2; ++ni) {
            int col = (nt0 + ni) * 16 + lane16;
            fb[ni] = fbias[p * E + col];
            gg[ni] = gam[col];
            bb[ni] = bet[col];
        }

        v8f acc[4][2];
        #pragma unroll
        for (int m = 0; m < 4; ++m) {
            // A fragments for this 16-row subtile (two b128 LDS loads each)
            FragU af[8];
            #pragma unroll
            for (int k = 0; k < 8; ++k) {
                const unsigned short* base =
                    &xs[(m * 16 + lane16) * XS_STRIDE + k * 32 + grp * 8];
                af[k].q[0] = *(const u32x4*)(base);
                af[k].q[1] = *(const u32x4*)(base + 16);
            }
            #pragma unroll
            for (int ni = 0; ni < 2; ++ni) {
                v8f a;
                #pragma unroll
                for (int r = 0; r < 8; ++r) a[r] = fb[ni];   // bias init of C
                #pragma unroll
                for (int k = 0; k < 8; ++k)
                    a = __builtin_amdgcn_wmma_f32_16x16x32_bf16(
                            false, af[k].v, false, bfr[ni][k].v,
                            (short)0, a, false, false);
                acc[m][ni] = a;
            }
            // residual add + block-wide LN statistics
            #pragma unroll
            for (int r = 0; r < 8; ++r) {
                int row_l = m * 16 + r + grp * 8;  // C layout: VGPR r -> M=r/r+8
                size_t rb = (size_t)(m0 + row_l) * E;
                float v0 = acc[m][0][r] + res[rb + nt0 * 16 + lane16];
                float v1 = acc[m][1][r] + res[rb + (nt0 + 1) * 16 + lane16];
                acc[m][0][r] = v0;
                acc[m][1][r] = v1;
                float s  = v0 + v1;
                float sq = v0 * v0 + v1 * v1;
                #pragma unroll
                for (int off = 8; off >= 1; off >>= 1) {
                    s  += __shfl_xor(s,  off, 16);
                    sq += __shfl_xor(sq, off, 16);
                }
                if (lane16 == 0) {
                    atomicAdd(&s_sum[row_l], s);    // ds_add_f32
                    atomicAdd(&s_ssq[row_l], sq);
                }
            }
        }
        __syncthreads();

        const float inv = 1.0f / (float)E;
        float* outp = out + (size_t)p * B_ROWS * E;
        #pragma unroll
        for (int m = 0; m < 4; ++m)
            #pragma unroll
            for (int r = 0; r < 8; ++r) {
                int row_l = m * 16 + r + grp * 8;
                float mean = s_sum[row_l] * inv;
                float var  = s_ssq[row_l] * inv - mean * mean;
                float rs   = rsqrtf(var + 1e-5f);
                size_t rb  = (size_t)(m0 + row_l) * E;
                #pragma unroll
                for (int ni = 0; ni < 2; ++ni) {
                    int col = (nt0 + ni) * 16 + lane16;
                    outp[rb + col] = (acc[m][ni][r] - mean) * rs * gg[ni] + bb[ni];
                }
            }
        __syncthreads();   // stats/xs reuse safe before next path
    }
}

extern "C" void kernel_launch(void* const* d_in, const int* in_sizes, int n_in,
                              void* d_out, int out_size, void* d_ws, size_t ws_size,
                              hipStream_t stream) {
    const float* audio     = (const float*)d_in[0];
    const float* text      = (const float*)d_in[1];
    const float* a2t_in_w  = (const float*)d_in[2];
    const float* a2t_in_b  = (const float*)d_in[3];
    const float* a2t_out_w = (const float*)d_in[4];
    const float* a2t_out_b = (const float*)d_in[5];
    const float* t2a_in_w  = (const float*)d_in[6];
    const float* t2a_in_b  = (const float*)d_in[7];
    const float* t2a_out_w = (const float*)d_in[8];
    const float* t2a_out_b = (const float*)d_in[9];
    const float* ln1_g     = (const float*)d_in[10];
    const float* ln1_b     = (const float*)d_in[11];
    const float* ln2_g     = (const float*)d_in[12];
    const float* ln2_b     = (const float*)d_in[13];
    float* out = (float*)d_out;

    unsigned short* packedW = (unsigned short*)d_ws;                    // 256 KB
    float* fbias = (float*)((char*)d_ws + (size_t)2 * 65536 * sizeof(unsigned short));

    pack_fused_w<<<512, 256, 0, stream>>>(a2t_in_w, a2t_out_w,
                                          t2a_in_w, t2a_out_w, packedW);
    fused_bias<<<2, 256, 0, stream>>>(a2t_in_b, a2t_out_w, a2t_out_b,
                                      t2a_in_b, t2a_out_w, t2a_out_b, fbias);

    dim3 grid(B_ROWS / M_TILE);
    cross_attn_main<<<grid, 256, 0, stream>>>(audio, text, packedW, fbias,
                                              ln1_g, ln1_b, ln2_g, ln2_b, out);
}
